// AdapterModel_63221918597613
// MI455X (gfx1250) — compile-verified, matrix operand order checked by reading, not run on hardware
//
#include <hip/hip_runtime.h>
#include <hip/hip_bf16.h>

// ---------------------------------------------------------------------------
// Types / constants
// ---------------------------------------------------------------------------
typedef __attribute__((ext_vector_type(16))) __bf16 v16bf;
typedef __attribute__((ext_vector_type(8)))  float  v8f;

#define NEGV (-1.0e9f)
#define LN_EPS (1e-5f)

// B=8, L=1024, H=1280, NH=20, HD=64, F=640, M=B*L=8192

__device__ __forceinline__ v8f wmma_bf16(v16bf a, v16bf b, v8f c) {
  return __builtin_amdgcn_wmma_f32_16x16x32_bf16(false, a, false, b,
                                                 (short)0, c, false, false);
}

// Async 16-byte global -> LDS copy (gfx1250 GLOBAL_LOAD_ASYNC_TO_LDS_B128,
// tracked by ASYNCcnt). LDS generic pointer's low 32 bits are the LDS byte
// address.
__device__ __forceinline__ void async_b128(void* lds, const void* gptr) {
  unsigned off32 = (unsigned)(unsigned long long)lds;
  asm volatile("global_load_async_to_lds_b128 %0, %1, off"
               :
               : "v"(off32), "v"(gptr)
               : "memory");
}
__device__ __forceinline__ void wait_async0() {
  asm volatile("s_wait_asynccnt 0" ::: "memory");
}

// A-fragment 16x32 (MxK), tile stored row-major: elem(m,k) = base[m*ld + k]
// lane&15 = M, lane>>4 selects K half; per-VGPR K mapping per ISA 7.12.2.
__device__ __forceinline__ v16bf load_frag_a(const __bf16* base, int ld, int lane) {
  const int m = lane & 15;
  const int half = lane >> 4;
  const __bf16* rp = base + m * ld + 8 * half;
  v16bf a;
#pragma unroll
  for (int e = 0; e < 16; ++e) {
    int kk = ((e & 8) << 1) + (e & 7);   // k - 8*half
    a[e] = rp[kk];
  }
  return a;
}

// B-fragment 32x16 (KxN), tile stored K-major: elem(k,n) = base[k*ld + n]
__device__ __forceinline__ v16bf load_frag_b_kn(const __bf16* base, int ld, int lane) {
  const int n = lane & 15;
  const int half = lane >> 4;
  const __bf16* cp = base + n + (8 * half) * ld;
  v16bf b;
#pragma unroll
  for (int e = 0; e < 16; ++e) {
    int kk = ((e & 8) << 1) + (e & 7);
    b[e] = cp[kk * ld];
  }
  return b;
}

// ---------------------------------------------------------------------------
// Small helper kernels
// ---------------------------------------------------------------------------
__global__ void cvt_f32_bf16_kernel(const float* __restrict__ in,
                                    __bf16* __restrict__ out, int n) {
  int i = blockIdx.x * 256 + threadIdx.x;
  if (i < n) out[i] = (__bf16)in[i];
}

// Xq = bf16(query_states); KV = bf16(emb_table[struct_ids])
__global__ void prep_kernel(const float* __restrict__ qs,
                            const int* __restrict__ sid,
                            const float* __restrict__ emb,
                            __bf16* __restrict__ Xq,
                            __bf16* __restrict__ KV) {
  int i = blockIdx.x * 256 + threadIdx.x;
  if (i >= 8192 * 1280) return;
  int col = i % 1280;
  int row = i / 1280;
  Xq[i] = (__bf16)qs[i];
  KV[i] = (__bf16)emb[(size_t)sid[row] * 1280 + col];
}

// ---------------------------------------------------------------------------
// GEMM: out[M,N] = A[M,K](bf16) @ W[K,N](bf16) + bias.
// 128x64 block tile, K-step 64, 8 waves (4M x 2N, 32x32 per wave),
// double-buffered async global->LDS pipeline. 8 WMMAs per wave per stage.
// ---------------------------------------------------------------------------
template <int OUT_BF16>
__global__ __launch_bounds__(256) void gemm_bf16_kernel(
    const __bf16* __restrict__ A, const __bf16* __restrict__ W,
    const float* __restrict__ bias, void* __restrict__ outp,
    int Mdim, int Ndim, int Kdim) {
  __shared__ __align__(16) __bf16 As[2][128 * 64];
  __shared__ __align__(16) __bf16 Bs[2][64 * 64];
  const int tid  = threadIdx.x;
  const int lane = tid & 31;
  const int wave = tid >> 5;
  const int wm   = wave & 3;         // 32-row sub-tile
  const int wn   = wave >> 2;        // 32-col sub-tile
  const int m0   = blockIdx.y * 128;
  const int n0g  = blockIdx.x * 64;

  v8f acc00 = {}, acc01 = {}, acc10 = {}, acc11 = {};

  const int nsteps = Kdim >> 6;

  // stage loader: 128x64 A chunk + 64x64 B chunk, 16B per lane per issue
  auto load_stage = [&](int buf, int k0) {
#pragma unroll
    for (int i = 0; i < 4; ++i) {           // 1024 chunks of 8 bf16
      int c = tid + 256 * i;
      int row = c >> 3, col = (c & 7) << 3;
      async_b128(&As[buf][row * 64 + col],
                 &A[(size_t)(m0 + row) * Kdim + k0 + col]);
    }
#pragma unroll
    for (int i = 0; i < 2; ++i) {           // 512 chunks
      int c = tid + 256 * i;
      int row = c >> 3, col = (c & 7) << 3;
      async_b128(&Bs[buf][row * 64 + col],
                 &W[(size_t)(k0 + row) * Ndim + n0g + col]);
    }
  };

  load_stage(0, 0);
  wait_async0();
  __syncthreads();

  for (int s = 0; s < nsteps; ++s) {
    const int buf = s & 1;
    if (s + 1 < nsteps) load_stage(buf ^ 1, (s + 1) << 6);

    const __bf16* Asb = As[buf];
    const __bf16* Bsb = Bs[buf];
#pragma unroll
    for (int kc = 0; kc < 2; ++kc) {
      v16bf a0 = load_frag_a(&Asb[(32 * wm) * 64 + 32 * kc], 64, lane);
      v16bf a1 = load_frag_a(&Asb[(32 * wm + 16) * 64 + 32 * kc], 64, lane);
      v16bf b0 = load_frag_b_kn(&Bsb[(32 * kc) * 64 + 32 * wn], 64, lane);
      v16bf b1 = load_frag_b_kn(&Bsb[(32 * kc) * 64 + 32 * wn + 16], 64, lane);
      acc00 = wmma_bf16(a0, b0, acc00);
      acc01 = wmma_bf16(a0, b1, acc01);
      acc10 = wmma_bf16(a1, b0, acc10);
      acc11 = wmma_bf16(a1, b1, acc11);
    }
    wait_async0();     // next stage resident in LDS
    __syncthreads();   // all waves done with current stage
  }

  const int half = lane >> 4, n15 = lane & 15;
#pragma unroll
  for (int mi = 0; mi < 2; ++mi) {
#pragma unroll
    for (int ni = 0; ni < 2; ++ni) {
      v8f acc = mi ? (ni ? acc11 : acc10) : (ni ? acc01 : acc00);
      int col = n0g + 32 * wn + 16 * ni + n15;
      float bvv = bias[col];
#pragma unroll
      for (int r = 0; r < 8; ++r) {
        int row = m0 + 32 * wm + 16 * mi + r + 8 * half;
        float val = acc[r] + bvv;
        if (OUT_BF16)
          ((__bf16*)outp)[(size_t)row * Ndim + col] = (__bf16)val;
        else
          ((float*)outp)[(size_t)row * Ndim + col] = val;
      }
    }
  }
}

// ---------------------------------------------------------------------------
// RoPE + head transpose + bf16 convert.
// In : Q/K/V fp32 [B*L, H];  Out: [B, NH, L, HD] bf16. Q scaled by HD^-0.5.
// ---------------------------------------------------------------------------
__global__ void rope_kernel(const float* __restrict__ Qf,
                            const float* __restrict__ Kf,
                            const float* __restrict__ Vf,
                            __bf16* __restrict__ Qh,
                            __bf16* __restrict__ Kh,
                            __bf16* __restrict__ Vh) {
  int idx = blockIdx.x * 256 + threadIdx.x;   // [b][h][l][d]
  if (idx >= 8 * 20 * 1024 * 64) return;
  int d  = idx & 63;
  int l  = (idx >> 6) & 1023;
  int bh = idx >> 16;
  int h  = bh % 20;
  int b  = bh / 20;
  size_t src = (size_t)(b * 1024 + l) * 1280 + h * 64;
  float ang = (float)l * __powf(10000.0f, -(float)(2 * (d & 31)) * (1.0f / 64.0f));
  float c = __cosf(ang), s = __sinf(ang);
  float qv = Qf[src + d];
  float qo = Qf[src + (d ^ 32)];
  float qr = (d < 32) ? -qo : qo;
  Qh[idx] = (__bf16)((qv * c + qr * s) * 0.125f);   // HD^-0.5 = 1/8
  float kv = Kf[src + d];
  float ko = Kf[src + (d ^ 32)];
  float kr = (d < 32) ? -ko : ko;
  Kh[idx] = (__bf16)(kv * c + kr * s);
  Vh[idx] = (__bf16)Vf[src + d];
}

// ---------------------------------------------------------------------------
// Flash attention: one block = 64 query rows of one (b,h); 4 waves x 16 rows.
// K/V tiles streamed via async global->LDS. ctx written as bf16 [B*L, H].
// ---------------------------------------------------------------------------
__global__ __launch_bounds__(128) void attn_kernel(
    const __bf16* __restrict__ Qh, const __bf16* __restrict__ Kh,
    const __bf16* __restrict__ Vh, const int* __restrict__ amask,
    __bf16* __restrict__ ctx) {
  __shared__ __align__(16) __bf16 Qs[64 * 64];
  __shared__ __align__(16) __bf16 Ks[64 * 64];
  __shared__ __align__(16) __bf16 Vs[64 * 64];
  __shared__ __align__(16) __bf16 Ps[4 * 16 * 64];  // per-wave P staging
  __shared__ float mskS[64];
  const int tid  = threadIdx.x;
  const int lane = tid & 31;
  const int w    = tid >> 5;
  const int half = lane >> 4, n15 = lane & 15;
  const int qt = blockIdx.x, h = blockIdx.y, b = blockIdx.z;
  const int qbase = qt * 64;
  const size_t headbase = (size_t)((b * 20 + h) * 1024) * 64;
  const __bf16* Qg = Qh + headbase + (size_t)qbase * 64;

  for (int i = tid; i < 512; i += 128) {
    int r = i >> 3, c = (i & 7) << 3;
    async_b128(&Qs[r * 64 + c], &Qg[(size_t)r * 64 + c]);
  }

  float row_m[8], row_l[8];
  v8f accO[4];
#pragma unroll
  for (int r = 0; r < 8; ++r) { row_m[r] = -1.0e30f; row_l[r] = 0.0f; }
#pragma unroll
  for (int t = 0; t < 4; ++t) accO[t] = (v8f){};

  for (int kt = 0; kt < 16; ++kt) {
    const int kbase = kt * 64;
    __syncthreads();   // prior iteration's compute done before overwriting K/V
    const __bf16* Kg = Kh + headbase + (size_t)kbase * 64;
    const __bf16* Vg = Vh + headbase + (size_t)kbase * 64;
    for (int i = tid; i < 512; i += 128) {
      int r = i >> 3, c = (i & 7) << 3;
      async_b128(&Ks[r * 64 + c], &Kg[(size_t)r * 64 + c]);
      async_b128(&Vs[r * 64 + c], &Vg[(size_t)r * 64 + c]);
    }
    if (tid < 64) mskS[tid] = (amask[b * 1024 + kbase + tid] == 0) ? 1.0f : 0.0f;
    wait_async0();
    __syncthreads();

    // S = Q @ K^T  (K tile stored [key][hd] => B in nk order == A layout)
    const __bf16* qrow = Qs + (16 * w) * 64;
    v16bf a0 = load_frag_a(qrow, 64, lane);
    v16bf a1 = load_frag_a(qrow + 32, 64, lane);
    v8f S[4];
#pragma unroll
    for (int t = 0; t < 4; ++t) {
      v8f s = {};
      s = wmma_bf16(a0, load_frag_a(Ks + (16 * t) * 64, 64, lane), s);
      s = wmma_bf16(a1, load_frag_a(Ks + (16 * t) * 64 + 32, 64, lane), s);
      float mflag = mskS[16 * t + n15];
#pragma unroll
      for (int r = 0; r < 8; ++r) s[r] = (mflag != 0.0f) ? NEGV : s[r];
      S[t] = s;
    }

    // online softmax (rows live within one 16-lane half)
    float tm[8];
#pragma unroll
    for (int r = 0; r < 8; ++r) {
      float v = fmaxf(fmaxf(S[0][r], S[1][r]), fmaxf(S[2][r], S[3][r]));
#pragma unroll
      for (int off = 8; off >= 1; off >>= 1) v = fmaxf(v, __shfl_xor(v, off, 32));
      tm[r] = v;
    }
    float alpha[8];
#pragma unroll
    for (int r = 0; r < 8; ++r) {
      float nm = fmaxf(row_m[r], tm[r]);
      alpha[r] = __expf(row_m[r] - nm);
      row_m[r] = nm;
    }
    __bf16* pw = Ps + w * 1024;
    float psum[8] = {0, 0, 0, 0, 0, 0, 0, 0};
#pragma unroll
    for (int t = 0; t < 4; ++t) {
#pragma unroll
      for (int r = 0; r < 8; ++r) {
        float p = __expf(S[t][r] - row_m[r]);
        psum[r] += p;
        pw[(r + 8 * half) * 64 + 16 * t + n15] = (__bf16)p;
      }
    }
#pragma unroll
    for (int r = 0; r < 8; ++r) {
      float v = psum[r];
#pragma unroll
      for (int off = 8; off >= 1; off >>= 1) v += __shfl_xor(v, off, 32);
      row_l[r] = row_l[r] * alpha[r] + v;
    }
#pragma unroll
    for (int t = 0; t < 4; ++t)
#pragma unroll
      for (int r = 0; r < 8; ++r) accO[t][r] *= alpha[r];

    // O += P @ V  (V tile [key][hd] is B in kn order)
#pragma unroll
    for (int kh = 0; kh < 2; ++kh) {
      v16bf ap = load_frag_a(pw + kh * 32, 64, lane);
#pragma unroll
      for (int t = 0; t < 4; ++t) {
        v16bf bv = load_frag_b_kn(Vs + (kh * 32) * 64 + 16 * t, 64, lane);
        accO[t] = wmma_bf16(ap, bv, accO[t]);
      }
    }
  }

  float invl[8];
#pragma unroll
  for (int r = 0; r < 8; ++r) invl[r] = 1.0f / row_l[r];
#pragma unroll
  for (int t = 0; t < 4; ++t) {
#pragma unroll
    for (int r = 0; r < 8; ++r) {
      int rg = qbase + 16 * w + r + 8 * half;
      int hd = 16 * t + n15;
      ctx[(size_t)(b * 1024 + rg) * 1280 + h * 64 + hd] =
          (__bf16)(accO[t][r] * invl[r]);
    }
  }
}

// ---------------------------------------------------------------------------
// Row-wise LayerNorm + ReLU, fp32 in -> bf16 out. One block per row.
// ---------------------------------------------------------------------------
__device__ __forceinline__ float block_reduce_sum256(float v, float* red) {
  int t = threadIdx.x;
  red[t] = v;
  __syncthreads();
#pragma unroll
  for (int s = 128; s > 0; s >>= 1) {
    if (t < s) red[t] += red[t + s];
    __syncthreads();
  }
  float r = red[0];
  __syncthreads();
  return r;
}

__device__ __forceinline__ float block_reduce_max256(float v, float* red) {
  int t = threadIdx.x;
  red[t] = v;
  __syncthreads();
#pragma unroll
  for (int s = 128; s > 0; s >>= 1) {
    if (t < s) red[t] = fmaxf(red[t], red[t + s]);
    __syncthreads();
  }
  float r = red[0];
  __syncthreads();
  return r;
}

__global__ __launch_bounds__(256) void ln_relu_kernel(
    const float* __restrict__ x, const float* __restrict__ g,
    const float* __restrict__ be, __bf16* __restrict__ out, int W) {
  __shared__ float red[256];
  const int row = blockIdx.x;
  const float* base = x + (size_t)row * W;
  float s1 = 0.0f, s2 = 0.0f;
  for (int c = threadIdx.x; c < W; c += 256) {
    float v = base[c];
    s1 += v;
    s2 += v * v;
  }
  s1 = block_reduce_sum256(s1, red);
  s2 = block_reduce_sum256(s2, red);
  float mean = s1 / (float)W;
  float var  = s2 / (float)W - mean * mean;
  float rstd = rsqrtf(var + LN_EPS);
  for (int c = threadIdx.x; c < W; c += 256) {
    float v = (base[c] - mean) * rstd * g[c] + be[c];
    out[(size_t)row * W + c] = (__bf16)fmaxf(v, 0.0f);
  }
}

// ---------------------------------------------------------------------------
// aw[task][bl] = tanh(T1[task][bl][:]) . pW2[task] + pb2[task]  (masked)
// One wave per (task, bl) row.
// ---------------------------------------------------------------------------
__global__ __launch_bounds__(256) void aw_kernel(
    const float* __restrict__ T1, const float* __restrict__ pW2,
    const float* __restrict__ pb2, const int* __restrict__ amask,
    float* __restrict__ aw) {
  int wid  = (blockIdx.x * 256 + threadIdx.x) >> 5;  // 0..24575
  int lane = threadIdx.x & 31;
  int task = wid / 8192;
  int bl   = wid % 8192;
  float acc = 0.0f;
  for (int c = lane; c < 320; c += 32)
    acc += tanhf(T1[(size_t)wid * 320 + c]) * pW2[task * 320 + c];
#pragma unroll
  for (int off = 16; off >= 1; off >>= 1) acc += __shfl_xor(acc, off, 32);
  if (lane == 0)
    aw[wid] = (amask[bl] == 0) ? NEGV : (acc + pb2[task]);
}

// ---------------------------------------------------------------------------
// softmax over L + pooled[tb][f] = sum_l f[b,l,f]*p[l]. One block per (task,b).
// ---------------------------------------------------------------------------
__global__ __launch_bounds__(256) void pool_kernel(
    const float* __restrict__ aw, const __bf16* __restrict__ fb,
    float* __restrict__ pooled) {
  __shared__ float p[1024];
  __shared__ float red[256];
  const int tb = blockIdx.x;   // task*8 + b
  const int task = tb >> 3, b = tb & 7;
  const float* arow = aw + task * 8192 + b * 1024;
  float lm = -1.0e30f;
  for (int l = threadIdx.x; l < 1024; l += 256) lm = fmaxf(lm, arow[l]);
  lm = block_reduce_max256(lm, red);
  float ls = 0.0f;
  for (int l = threadIdx.x; l < 1024; l += 256) {
    float e = __expf(arow[l] - lm);
    p[l] = e;
    ls += e;
  }
  ls = block_reduce_sum256(ls, red);
  float rs = 1.0f / ls;
  __syncthreads();
  for (int ff = threadIdx.x; ff < 640; ff += 256) {
    float acc = 0.0f;
    for (int l = 0; l < 1024; ++l)
      acc += (float)fb[(size_t)(b * 1024 + l) * 640 + ff] * p[l];
    pooled[tb * 640 + ff] = acc * rs;
  }
}

// ---------------------------------------------------------------------------
// Regression head: h=relu(LN(pooled@rW1+rb1)); h=relu(h@rW2+rb2); out=h@rW3+rb3
// One block per (task, b).
// ---------------------------------------------------------------------------
__global__ __launch_bounds__(256) void head_kernel(
    const float* __restrict__ pooled, const float* __restrict__ rW1,
    const float* __restrict__ rb1, const float* __restrict__ rg1,
    const float* __restrict__ rbe1, const float* __restrict__ rW2,
    const float* __restrict__ rb2, const float* __restrict__ rW3,
    const float* __restrict__ rb3, float* __restrict__ out) {
  __shared__ float pl[640];
  __shared__ float pre[320];
  __shared__ float h1s[320];
  __shared__ float h2s[160];
  __shared__ float red[256];
  const int tb = blockIdx.x;
  const int task = tb >> 3, b = tb & 7;
  const int tid = threadIdx.x;
  for (int ff = tid; ff < 640; ff += 256) pl[ff] = pooled[tb * 640 + ff];
  __syncthreads();
  for (int j = tid; j < 320; j += 256) {
    float acc = rb1[task * 320 + j];
    for (int f = 0; f < 640; ++f)
      acc += pl[f] * rW1[(size_t)(task * 640 + f) * 320 + j];
    pre[j] = acc;
  }
  __syncthreads();
  float s1 = 0.0f, s2 = 0.0f;
  for (int j = tid; j < 320; j += 256) {
    float v = pre[j];
    s1 += v;
    s2 += v * v;
  }
  s1 = block_reduce_sum256(s1, red);
  s2 = block_reduce_sum256(s2, red);
  float mean = s1 / 320.0f;
  float var  = s2 / 320.0f - mean * mean;
  float rstd = rsqrtf(var + LN_EPS);
  for (int j = tid; j < 320; j += 256) {
    float v = (pre[j] - mean) * rstd * rg1[task * 320 + j] + rbe1[task * 320 + j];
    h1s[j] = fmaxf(v, 0.0f);
  }
  __syncthreads();
  if (tid < 160) {
    float acc = rb2[task * 160 + tid];
    for (int j = 0; j < 320; ++j)
      acc += h1s[j] * rW2[(size_t)(task * 320 + j) * 160 + tid];
    h2s[tid] = fmaxf(acc, 0.0f);
  }
  __syncthreads();
  float part = (tid < 160) ? h2s[tid] * rW3[task * 160 + tid] : 0.0f;
  part = block_reduce_sum256(part, red);
  if (tid == 0) out[b * 3 + task] = part + rb3[task];
}

// ---------------------------------------------------------------------------
// Host orchestration
// ---------------------------------------------------------------------------
extern "C" void kernel_launch(void* const* d_in, const int* in_sizes, int n_in,
                              void* d_out, int out_size, void* d_ws, size_t ws_size,
                              hipStream_t stream) {
  (void)in_sizes; (void)n_in; (void)out_size; (void)ws_size;
  const float* qs    = (const float*)d_in[0];
  const int*   sid   = (const int*)d_in[1];
  const int*   amask = (const int*)d_in[2];
  const float* emb   = (const float*)d_in[3];
  const float* Wq = (const float*)d_in[4];   const float* bq = (const float*)d_in[5];
  const float* Wk = (const float*)d_in[6];   const float* bk = (const float*)d_in[7];
  const float* Wv = (const float*)d_in[8];   const float* bv = (const float*)d_in[9];
  const float* Wo = (const float*)d_in[10];  const float* bo = (const float*)d_in[11];
  const float* W1 = (const float*)d_in[12];  const float* b1 = (const float*)d_in[13];
  const float* g1 = (const float*)d_in[14];  const float* be1 = (const float*)d_in[15];
  const float* W2 = (const float*)d_in[16];  const float* b2 = (const float*)d_in[17];
  const float* g2 = (const float*)d_in[18];  const float* be2 = (const float*)d_in[19];
  const float* pW1 = (const float*)d_in[20]; const float* pb1 = (const float*)d_in[21];
  const float* pW2 = (const float*)d_in[22]; const float* pb2 = (const float*)d_in[23];
  const float* rW1 = (const float*)d_in[24]; const float* rb1 = (const float*)d_in[25];
  const float* rg1 = (const float*)d_in[26]; const float* rbe1 = (const float*)d_in[27];
  const float* rW2 = (const float*)d_in[28]; const float* rb2 = (const float*)d_in[29];
  const float* rW3 = (const float*)d_in[30]; const float* rb3 = (const float*)d_in[31];
  float* out = (float*)d_out;
  char* ws = (char*)d_ws;

  const size_t Mrows = 8192;
  size_t off = 0;
  auto alloc = [&](size_t bytes) {
    size_t cur = off;
    off += (bytes + 255) & ~(size_t)255;
    return cur;
  };
  size_t oXq  = alloc(Mrows * 1280 * 2);
  size_t oKV  = alloc(Mrows * 1280 * 2);
  size_t oWqb = alloc(1280 * 1280 * 2);
  size_t oWkb = alloc(1280 * 1280 * 2);
  size_t oWvb = alloc(1280 * 1280 * 2);
  size_t oWob = alloc(1280 * 1280 * 2);
  size_t oW1b = alloc(1280 * 1280 * 2);
  size_t oW2b = alloc(1280 * 640 * 2);
  size_t opW1b = alloc(3 * 640 * 320 * 2);
  size_t oQ   = alloc(Mrows * 1280 * 4);
  size_t oK   = alloc(Mrows * 1280 * 4);
  size_t oV   = alloc(Mrows * 1280 * 4);
  size_t oQh  = alloc(Mrows * 1280 * 2);
  size_t oKh  = alloc(Mrows * 1280 * 2);
  size_t oVh  = alloc(Mrows * 1280 * 2);
  // aliases (lifetimes disjoint):
  size_t oCtx = oXq;           // ctx bf16 after Xq consumed
  size_t oAtt = oKV;           // attn_out bf16 after KV consumed
  size_t oY1  = oQ;            // fp32, after Q->Qh
  size_t oF1  = oQh;           // f1 bf16, after attention
  size_t oY2  = oK;            // fp32 [8192,640]
  size_t oF   = oKh;           // f bf16 [8192,640]
  size_t oT1  = oV;            // fp32 [3,8192,320]
  size_t oAw  = oVh;           // fp32 [3,8192]
  size_t oPool = oVh + 98304;  // fp32 [3,8,640]

  __bf16* Xq   = (__bf16*)(ws + oXq);
  __bf16* KV   = (__bf16*)(ws + oKV);
  __bf16* Wqb  = (__bf16*)(ws + oWqb);
  __bf16* Wkb  = (__bf16*)(ws + oWkb);
  __bf16* Wvb  = (__bf16*)(ws + oWvb);
  __bf16* Wob  = (__bf16*)(ws + oWob);
  __bf16* W1b  = (__bf16*)(ws + oW1b);
  __bf16* W2b  = (__bf16*)(ws + oW2b);
  __bf16* pW1b = (__bf16*)(ws + opW1b);
  float*  Qf   = (float*)(ws + oQ);
  float*  Kf   = (float*)(ws + oK);
  float*  Vf   = (float*)(ws + oV);
  __bf16* Qh   = (__bf16*)(ws + oQh);
  __bf16* Kh   = (__bf16*)(ws + oKh);
  __bf16* Vh   = (__bf16*)(ws + oVh);
  __bf16* Ctx  = (__bf16*)(ws + oCtx);
  __bf16* Att  = (__bf16*)(ws + oAtt);
  float*  Y1   = (float*)(ws + oY1);
  __bf16* F1   = (__bf16*)(ws + oF1);
  float*  Y2   = (float*)(ws + oY2);
  __bf16* Fb   = (__bf16*)(ws + oF);
  float*  T1   = (float*)(ws + oT1);
  float*  Aw   = (float*)(ws + oAw);
  float*  Pool = (float*)(ws + oPool);

  // 1) convert weights to bf16
  cvt_f32_bf16_kernel<<<(1638400 + 255) / 256, 256, 0, stream>>>(Wq, Wqb, 1638400);
  cvt_f32_bf16_kernel<<<(1638400 + 255) / 256, 256, 0, stream>>>(Wk, Wkb, 1638400);
  cvt_f32_bf16_kernel<<<(1638400 + 255) / 256, 256, 0, stream>>>(Wv, Wvb, 1638400);
  cvt_f32_bf16_kernel<<<(1638400 + 255) / 256, 256, 0, stream>>>(Wo, Wob, 1638400);
  cvt_f32_bf16_kernel<<<(1638400 + 255) / 256, 256, 0, stream>>>(W1, W1b, 1638400);
  cvt_f32_bf16_kernel<<<(819200 + 255) / 256, 256, 0, stream>>>(W2, W2b, 819200);
  cvt_f32_bf16_kernel<<<(614400 + 255) / 256, 256, 0, stream>>>(pW1, pW1b, 614400);

  // 2) activations: bf16 query, gathered struct embeddings
  prep_kernel<<<40960, 256, 0, stream>>>(qs, sid, emb, Xq, KV);

  // 3) Q/K/V projections (WMMA bf16, fp32 out)
  gemm_bf16_kernel<0><<<dim3(20, 64), 256, 0, stream>>>(Xq, Wqb, bq, Qf, 8192, 1280, 1280);
  gemm_bf16_kernel<0><<<dim3(20, 64), 256, 0, stream>>>(KV, Wkb, bk, Kf, 8192, 1280, 1280);
  gemm_bf16_kernel<0><<<dim3(20, 64), 256, 0, stream>>>(KV, Wvb, bv, Vf, 8192, 1280, 1280);

  // 4) RoPE + scale + head layout + bf16
  rope_kernel<<<40960, 256, 0, stream>>>(Qf, Kf, Vf, Qh, Kh, Vh);

  // 5) flash attention (WMMA QK^T and PV), ctx bf16 [B*L,H]
  attn_kernel<<<dim3(16, 20, 8), 128, 0, stream>>>(Qh, Kh, Vh, amask, Ctx);

  // 6) output proj (bf16 out), W1, LN+ReLU, W2, LN+ReLU
  gemm_bf16_kernel<1><<<dim3(20, 64), 256, 0, stream>>>(Ctx, Wob, bo, Att, 8192, 1280, 1280);
  gemm_bf16_kernel<0><<<dim3(20, 64), 256, 0, stream>>>(Att, W1b, b1, Y1, 8192, 1280, 1280);
  ln_relu_kernel<<<8192, 256, 0, stream>>>(Y1, g1, be1, F1, 1280);
  gemm_bf16_kernel<0><<<dim3(10, 64), 256, 0, stream>>>(F1, W2b, b2, Y2, 8192, 640, 1280);
  ln_relu_kernel<<<8192, 256, 0, stream>>>(Y2, g2, be2, Fb, 640);

  // 7) per-task pooling attention: T1 = f @ pW1 + pb1 (WMMA), then tanh-dot
  for (int t = 0; t < 3; ++t) {
    gemm_bf16_kernel<0><<<dim3(5, 64), 256, 0, stream>>>(
        Fb, pW1b + (size_t)t * 640 * 320, pb1 + t * 320,
        T1 + (size_t)t * 8192 * 320, 8192, 320, 640);
  }
  aw_kernel<<<3072, 256, 0, stream>>>(T1, pW2, pb2, amask, Aw);
  pool_kernel<<<24, 256, 0, stream>>>(Aw, Fb, Pool);
  head_kernel<<<24, 256, 0, stream>>>(Pool, rW1, rb1, rg1, rbe1, rW2, rb2, rW3, rb3, out);
}
